// ParallelDroplessMLP_2302102471530
// MI455X (gfx1250) — matrix operand order repaired and checked by reading, not compiled, not run on hardware
//
#include <hip/hip_runtime.h>

// ---------------------------------------------------------------------------
// MoE dropless MLP for MI455X (gfx1250): bf16 WMMA grouped GEMM.
// Wave tile: 32 rows x 64 cols (8 accumulators), branch-free inner loop.
// ---------------------------------------------------------------------------

#define E_      8
#define TOPK_   2
#define SL_     2048
#define HS_     1024
#define FFN_    4096
#define TRIM_   256                       // SL/E
#define NA_     5632                      // 256*8 + 1792*2 assignments
#define NPAD_   5888                      // NA_ padded so each expert seg is %32
#define TROWS_  32                        // rows per tile
#define NTILE_  184                       // NPAD_/TROWS_ row tiles (static grid)

typedef __attribute__((ext_vector_type(16))) __bf16 v16bf;
typedef __attribute__((ext_vector_type(8)))  float  v8f;

union Frag { v16bf v; uint4 q[2]; };

__device__ inline v16bf load_frag(const __bf16* p0, const __bf16* p1) {
  Frag f;
  f.q[0] = *(const uint4*)p0;   // 8 bf16
  f.q[1] = *(const uint4*)p1;   // 8 bf16
  return f.v;
}

__device__ inline float gelu_f(float v) {
  return 0.5f * v * (1.0f + erff(v * 0.70710678118654752f));
}

#define WMMA_BF16(ACC, A, B) \
  ACC = __builtin_amdgcn_wmma_f32_16x16x32_bf16(false, A, false, B, (short)0, ACC, false, false)

// --- conversions ------------------------------------------------------------

__global__ void conv_x_kernel(const float* __restrict__ x, __bf16* __restrict__ xb) {
  size_t i = (size_t)blockIdx.x * 256 + threadIdx.x;
  if (i < (size_t)SL_ * HS_) xb[i] = (__bf16)x[i];
}

// w [E][K][N] fp32 row-major -> ws [E][K/16][N][16] bf16 (B-fragment layout)
__global__ void swizzle_w_kernel(const float* __restrict__ w, __bf16* __restrict__ ws_,
                                 int ksize, int nsize) {
  size_t tid = (size_t)blockIdx.x * 256 + threadIdx.x;
  size_t total = (size_t)E_ * ksize * nsize;
  if (tid >= total) return;
  int kk = (int)(tid & 15);
  size_t r = tid >> 4;
  int n  = (int)(r % nsize); r /= nsize;
  int kchunks = ksize >> 4;
  int kc = (int)(r % kchunks);
  int e  = (int)(r / kchunks);
  int k  = kc * 16 + kk;
  ws_[tid] = (__bf16)w[((size_t)e * ksize + k) * nsize + n];
}

// --- routing ----------------------------------------------------------------

__global__ void routing_kernel(const float* __restrict__ scores,
                               const float* __restrict__ ew,
                               const int*   __restrict__ ei,
                               int* __restrict__ atok, int* __restrict__ aexp,
                               float* __restrict__ awt) {
  int a = blockIdx.x * 256 + threadIdx.x;
  if (a >= NA_) return;
  int t, e; float w;
  if (a < TRIM_ * E_) {                       // dense prefix: every expert
    t = a >> 3; e = a & 7; w = scores[t * E_ + e];
  } else {                                    // sparse top-k
    int s = a - TRIM_ * E_;
    t = TRIM_ + (s >> 1);
    int j = s & 1;
    e = ei[t * TOPK_ + j];
    w = ew[t * TOPK_ + j];
  }
  atok[a] = t; aexp[a] = e; awt[a] = w;
}

// deterministic counting sort by expert, padded per-expert to TROWS_ rows
__global__ void sort_kernel(const int* __restrict__ aexp, const int* __restrict__ atok,
                            int* __restrict__ stok, int* __restrict__ sorig,
                            int* __restrict__ texp) {
  if (threadIdx.x != 0 || blockIdx.x != 0) return;
  int cnt[E_];
  for (int e = 0; e < E_; ++e) cnt[e] = 0;
  for (int a = 0; a < NA_; ++a) cnt[aexp[a]]++;
  int base[E_ + 1]; int off = 0;
  for (int e = 0; e < E_; ++e) { base[e] = off; off += ((cnt[e] + TROWS_ - 1) / TROWS_) * TROWS_; }
  base[E_] = off;
  for (int i = 0; i < NPAD_; ++i) { stok[i] = -1; sorig[i] = -1; }
  int pos[E_];
  for (int e = 0; e < E_; ++e) pos[e] = base[e];
  for (int a = 0; a < NA_; ++a) {
    int e = aexp[a]; int p = pos[e]++;
    stok[p] = atok[a]; sorig[p] = a;
  }
  int te = 0;
  for (int i = 0; i < NTILE_; ++i) {
    while (te < E_ - 1 && i * TROWS_ >= base[te + 1]) te++;
    texp[i] = te;
  }
}

// --- grouped GEMM 1:  h[srow, FFN] = gelu( xb[token] @ w1s[e] ) -------------
// block = 4 waves; wave covers 32 rows x 64 cols (8 wmma accumulators).
// Padding rows are clamped to token 0: their outputs are garbage but only ever
// feed padding rows of GEMM2, which are discarded (row m of D depends only on
// row m of A). This keeps the inner loop branch-free.

__global__ __launch_bounds__(128) void gemm1_kernel(
    const __bf16* __restrict__ xb, const __bf16* __restrict__ w1s,
    const int* __restrict__ stok, const int* __restrict__ texp,
    __bf16* __restrict__ h) {
  const int wave = threadIdx.x >> 5;
  const int lane = threadIdx.x & 31;
  const int half = lane >> 4;
  const int l16  = lane & 15;
  const int tile = blockIdx.y;
  const int e    = texp[tile];
  const int srow0 = tile * TROWS_;
  const int n0 = blockIdx.x * 256 + wave * 64;

  int t0 = stok[srow0 + l16];       if (t0 < 0) t0 = 0;
  int t1 = stok[srow0 + 16 + l16];  if (t1 < 0) t1 = 0;
  const __bf16* arow0 = xb + (size_t)t0 * HS_;
  const __bf16* arow1 = xb + (size_t)t1 * HS_;
  const __bf16* wbase = w1s + (size_t)e * (HS_ / 16) * FFN_ * 16;

  v8f acc00 = {}, acc01 = {}, acc02 = {}, acc03 = {};
  v8f acc10 = {}, acc11 = {}, acc12 = {}, acc13 = {};
  for (int k0 = 0; k0 < HS_; k0 += 32) {
    v16bf a0 = load_frag(arow0 + k0 + half * 16, arow0 + k0 + half * 16 + 8);
    v16bf a1 = load_frag(arow1 + k0 + half * 16, arow1 + k0 + half * 16 + 8);
    const int kc = (k0 >> 4) + half;
    const __bf16* bcol = wbase + ((size_t)kc * FFN_ + n0 + l16) * 16;
    v16bf b0 = load_frag(bcol +   0, bcol +   8);
    v16bf b1 = load_frag(bcol + 256, bcol + 264);
    v16bf b2 = load_frag(bcol + 512, bcol + 520);
    v16bf b3 = load_frag(bcol + 768, bcol + 776);
    WMMA_BF16(acc00, a0, b0);
    WMMA_BF16(acc01, a0, b1);
    WMMA_BF16(acc02, a0, b2);
    WMMA_BF16(acc03, a0, b3);
    WMMA_BF16(acc10, a1, b0);
    WMMA_BF16(acc11, a1, b1);
    WMMA_BF16(acc12, a1, b2);
    WMMA_BF16(acc13, a1, b3);
  }
#pragma unroll
  for (int v = 0; v < 8; ++v) {
    int rowA = srow0 + v + half * 8;
    size_t baseA = (size_t)rowA * FFN_ + n0 + l16;
    h[baseA +  0] = (__bf16)gelu_f(acc00[v]);
    h[baseA + 16] = (__bf16)gelu_f(acc01[v]);
    h[baseA + 32] = (__bf16)gelu_f(acc02[v]);
    h[baseA + 48] = (__bf16)gelu_f(acc03[v]);
    size_t baseB = baseA + (size_t)16 * FFN_;
    h[baseB +  0] = (__bf16)gelu_f(acc10[v]);
    h[baseB + 16] = (__bf16)gelu_f(acc11[v]);
    h[baseB + 32] = (__bf16)gelu_f(acc12[v]);
    h[baseB + 48] = (__bf16)gelu_f(acc13[v]);
  }
}

// --- grouped GEMM 2:  y[orig, HS] = h[srow] @ w2s[e] ------------------------

__global__ __launch_bounds__(128) void gemm2_kernel(
    const __bf16* __restrict__ h, const __bf16* __restrict__ w2s,
    const int* __restrict__ sorig, const int* __restrict__ texp,
    float* __restrict__ y) {
  const int wave = threadIdx.x >> 5;
  const int lane = threadIdx.x & 31;
  const int half = lane >> 4;
  const int l16  = lane & 15;
  const int tile = blockIdx.y;
  const int e    = texp[tile];
  const int srow0 = tile * TROWS_;
  const int n0 = blockIdx.x * 256 + wave * 64;

  const __bf16* arow0 = h + (size_t)(srow0 + l16) * FFN_;
  const __bf16* arow1 = h + (size_t)(srow0 + 16 + l16) * FFN_;
  const __bf16* wbase = w2s + (size_t)e * (FFN_ / 16) * HS_ * 16;

  v8f acc00 = {}, acc01 = {}, acc02 = {}, acc03 = {};
  v8f acc10 = {}, acc11 = {}, acc12 = {}, acc13 = {};
  for (int k0 = 0; k0 < FFN_; k0 += 32) {
    v16bf a0 = load_frag(arow0 + k0 + half * 16, arow0 + k0 + half * 16 + 8);
    v16bf a1 = load_frag(arow1 + k0 + half * 16, arow1 + k0 + half * 16 + 8);
    const int kc = (k0 >> 4) + half;
    const __bf16* bcol = wbase + ((size_t)kc * HS_ + n0 + l16) * 16;
    v16bf b0 = load_frag(bcol +   0, bcol +   8);
    v16bf b1 = load_frag(bcol + 256, bcol + 264);
    v16bf b2 = load_frag(bcol + 512, bcol + 520);
    v16bf b3 = load_frag(bcol + 768, bcol + 776);
    WMMA_BF16(acc00, a0, b0);
    WMMA_BF16(acc01, a0, b1);
    WMMA_BF16(acc02, a0, b2);
    WMMA_BF16(acc03, a0, b3);
    WMMA_BF16(acc10, a1, b0);
    WMMA_BF16(acc11, a1, b1);
    WMMA_BF16(acc12, a1, b2);
    WMMA_BF16(acc13, a1, b3);
  }
#pragma unroll
  for (int v = 0; v < 8; ++v) {
    int rowA = srow0 + v + half * 8;
    int origA = sorig[rowA];
    if (origA >= 0) {
      size_t base = (size_t)origA * HS_ + n0 + l16;
      y[base +  0] = acc00[v];
      y[base + 16] = acc01[v];
      y[base + 32] = acc02[v];
      y[base + 48] = acc03[v];
    }
    int origB = sorig[rowA + 16];
    if (origB >= 0) {
      size_t base = (size_t)origB * HS_ + n0 + l16;
      y[base +  0] = acc10[v];
      y[base + 16] = acc11[v];
      y[base + 32] = acc12[v];
      y[base + 48] = acc13[v];
    }
  }
}

// --- combine: weighted sum -> out ; unweighted scatter -> buffer ------------

__global__ void combine_kernel(const float* __restrict__ y,
                               const float* __restrict__ awt,
                               const int*   __restrict__ aexp,
                               float* __restrict__ out, float* __restrict__ buf) {
  int tid = blockIdx.x * 256 + threadIdx.x;
  if (tid >= SL_ * HS_) return;
  int t = tid >> 10;          // token
  int c = tid & (HS_ - 1);    // column
  float o = 0.f;
  float b[E_] = {0.f, 0.f, 0.f, 0.f, 0.f, 0.f, 0.f, 0.f};
  if (t < TRIM_) {
#pragma unroll
    for (int j = 0; j < E_; ++j) {
      int a = t * E_ + j;
      float yv = y[(size_t)a * HS_ + c];
      o += awt[a] * yv;
      b[j] += yv;                            // dense: expert == j
    }
  } else {
    for (int j = 0; j < TOPK_; ++j) {
      int a = TRIM_ * E_ + (t - TRIM_) * TOPK_ + j;
      float yv = y[(size_t)a * HS_ + c];
      o += awt[a] * yv;
      int e = aexp[a];
#pragma unroll
      for (int q = 0; q < E_; ++q) if (q == e) b[q] += yv;
    }
  }
  out[tid] = o;
  float* bp = buf + (size_t)t * E_ * HS_ + c;
#pragma unroll
  for (int q = 0; q < E_; ++q) bp[(size_t)q * HS_] = b[q];
}

// ---------------------------------------------------------------------------

extern "C" void kernel_launch(void* const* d_in, const int* in_sizes, int n_in,
                              void* d_out, int out_size, void* d_ws, size_t ws_size,
                              hipStream_t stream) {
  const float* x  = (const float*)d_in[0];
  const float* ew = (const float*)d_in[1];
  const int*   ei = (const int*)d_in[2];
  const float* sc = (const float*)d_in[3];
  const float* w1 = (const float*)d_in[4];
  const float* w2 = (const float*)d_in[5];
  float* out = (float*)d_out;
  float* buf = out + (size_t)SL_ * HS_;

  char* ws = (char*)d_ws;
  size_t off = 0;
  auto carve = [&](size_t bytes) -> void* {
    void* p = ws + off;
    off = (off + bytes + 255) & ~(size_t)255;
    return p;
  };
  __bf16* xb   = (__bf16*)carve((size_t)SL_ * HS_ * 2);
  __bf16* w1s  = (__bf16*)carve((size_t)E_ * HS_ * FFN_ * 2);
  __bf16* w2s  = (__bf16*)carve((size_t)E_ * FFN_ * HS_ * 2);
  __bf16* hbuf = (__bf16*)carve((size_t)NPAD_ * FFN_ * 2);
  float*  y    = (float*)carve((size_t)NA_ * HS_ * 4);
  int*    atok = (int*)carve(NA_ * 4);
  int*    aexp = (int*)carve(NA_ * 4);
  float*  awt  = (float*)carve(NA_ * 4);
  int*    stok = (int*)carve(NPAD_ * 4);
  int*    sorig= (int*)carve(NPAD_ * 4);
  int*    texp = (int*)carve(NTILE_ * 4);

  conv_x_kernel<<<(SL_ * HS_ + 255) / 256, 256, 0, stream>>>(x, xb);
  swizzle_w_kernel<<<(int)(((size_t)E_ * HS_ * FFN_) / 256), 256, 0, stream>>>(w1, w1s, HS_, FFN_);
  swizzle_w_kernel<<<(int)(((size_t)E_ * FFN_ * HS_) / 256), 256, 0, stream>>>(w2, w2s, FFN_, HS_);
  routing_kernel<<<(NA_ + 255) / 256, 256, 0, stream>>>(sc, ew, ei, atok, aexp, awt);
  sort_kernel<<<1, 1, 0, stream>>>(aexp, atok, stok, sorig, texp);
  gemm1_kernel<<<dim3(FFN_ / 256, NTILE_), 128, 0, stream>>>(xb, w1s, stok, texp, hbuf);
  gemm2_kernel<<<dim3(HS_ / 256, NTILE_), 128, 0, stream>>>(hbuf, w2s, sorig, texp, y);
  combine_kernel<<<(SL_ * HS_ + 255) / 256, 256, 0, stream>>>(y, awt, aexp, out, buf);
}
